// ConformalMIL_6287832121415
// MI455X (gfx1250) — matrix operand order repaired and checked by reading, not compiled, not run on hardware
//
#include <hip/hip_runtime.h>
#include <stdint.h>

#define USE_ASYNC 1

// ---------------------------------------------------------------------------
// Shapes
static constexpr int Bsz  = 32;
static constexpr int S    = 2048;
static constexpr int D    = 512;
static constexpr int NCLS = 2;
static constexpr int H    = 16;
static constexpr int HD   = 32;
static constexpr int M    = Bsz * S;      // 65536 token rows

// ---------------------------------------------------------------------------
// WMMA types (gfx1250 / CDNA5)
typedef __attribute__((ext_vector_type(16))) __bf16 v16bf;
typedef __attribute__((ext_vector_type(8)))  float  v8f;

union FragBF { uint32_t u[8]; v16bf v; };

// f32 -> bf16 round-to-nearest-even
__device__ __forceinline__ uint16_t f2bf(float f) {
    union { float f; uint32_t u; } x; x.f = f;
    uint32_t r = x.u + 0x7FFFu + ((x.u >> 16) & 1u);
    return (uint16_t)(r >> 16);
}
__device__ __forceinline__ float bf2f(uint16_t h) {
    union { uint32_t u; float f; } x; x.u = ((uint32_t)h) << 16;
    return x.f;
}

// ---------------------------------------------------------------------------
// 1) Mexican hat wavelet kernels, summed over the 3 wave params: Kc[D][19]
__global__ void wavelet_kernel(const float* __restrict__ w1,
                               const float* __restrict__ w2,
                               const float* __restrict__ w3,
                               float* __restrict__ kc) {
    int idx = blockIdx.x * 256 + threadIdx.x;
    if (idx >= D * 19) return;
    int d = idx / 19, t = idx % 19;
    float xs = (float)(t - 9);
    const float C = 0.86732507058f;   // 2 / (sqrt(3) * pi^0.25)
    const float* wv[3] = { w1, w2, w3 };
    float acc = 0.f;
#pragma unroll
    for (int i = 0; i < 3; ++i) {
        float scale = wv[i][d];          // wave[0][d][0]
        float shift = wv[i][D + d];      // wave[1][d][0]
        float u = (xs - shift) / scale;
        acc += C * (1.f - u * u) * expf(-0.5f * u * u) * rsqrtf(fabsf(scale));
    }
    kc[idx] = acc;
}

// ---------------------------------------------------------------------------
// 2) Pack f32 weight W[K][N] into bf16 B-fragment layout:
//    out[((kt*NT + nt)*32 + lane)*8 + v] = {W[k][n], W[k+1][n]} packed,
//    n = nt*16 + (lane&15), k = kt*32 + (lane<16?0:8) + (v>=4?16:0) + 2(v&3)   (ISA 7.12.2)
__global__ void pack_b_kernel(const float* __restrict__ W, uint32_t* __restrict__ out,
                              int K, int N) {
    int idx = blockIdx.x * 256 + threadIdx.x;
    int total = (K / 2) * N;
    if (idx >= total) return;
    int v    = idx & 7;
    int lane = (idx >> 3) & 31;
    int rest = idx >> 8;
    int NT = N >> 4;
    int nt = rest % NT;
    int kt = rest / NT;
    int n = nt * 16 + (lane & 15);
    int k = kt * 32 + ((lane >> 4) << 3) + ((v & 4) << 2) + ((v & 3) << 1);
    uint32_t lo = f2bf(W[(size_t)k * N + n]);
    uint32_t hi = f2bf(W[(size_t)(k + 1) * N + n]);
    out[idx] = lo | (hi << 16);
}

// ---------------------------------------------------------------------------
// 3) Depthwise 'same' conv, KSIZE=19, rolling register window per (b,d).
//    Output pos in bf16 row-major [B*S][D]  (A-operand friendly).
__global__ void dwconv_kernel(const float* __restrict__ x,
                              const float* __restrict__ kc,
                              uint16_t* __restrict__ pos) {
    int bid = blockIdx.x;           // 256 blocks: b(32) x dc(2) x sc(4)
    int sc = bid & 3;
    int dc = (bid >> 2) & 1;
    int b  = bid >> 3;
    int d  = dc * 256 + threadIdx.x;
    int s0 = sc * 512;
    float k[19];
#pragma unroll
    for (int i = 0; i < 19; ++i) k[i] = kc[d * 19 + i];
    const float* xb = x + (size_t)b * S * D + d;
    uint16_t* pb = pos + (size_t)b * S * D + d;
    float w[19];
#pragma unroll
    for (int i = 0; i < 19; ++i) {
        int s = s0 - 9 + i;
        w[i] = (s >= 0 && s < S) ? xb[(size_t)s * D] : 0.f;
    }
    for (int s = s0; s < s0 + 512; ++s) {
        float acc = 0.f;
#pragma unroll
        for (int i = 0; i < 19; ++i) acc += w[i] * k[i];
        pb[(size_t)s * D] = f2bf(acc);
#pragma unroll
        for (int i = 0; i < 18; ++i) w[i] = w[i + 1];
        int sn = s + 10;
        w[18] = (sn < S) ? xb[(size_t)sn * D] : 0.f;
    }
}

// ---------------------------------------------------------------------------
// 4/6) WMMA GEMM: C[M][N] = A[M][512] @ B[512][N],  K = 512.
//   A: bf16 row-major (global b128 loads); B: pre-packed fragments staged
//   once per block into LDS via global_load_async_to_lds_b128 (ASYNCcnt).
//   One wave -> 32 rows x 64 cols = 8 accumulators, 128 v_wmma per wave.
//   MODE 0: outf = resid + C + bias (f32, N=512)
//   MODE 1: outb = bf16(C)          (N=1024)
template <int MODE>
__global__ void __launch_bounds__(256)
wmma_gemm_kernel(const uint16_t* __restrict__ A,
                 const uint32_t* __restrict__ Bp,
                 const float* __restrict__ resid,
                 const float* __restrict__ bias,
                 float* __restrict__ outf,
                 uint16_t* __restrict__ outb,
                 int N) {
    const int K  = 512;
    const int KT = K / 32;                       // 16 k-steps
    __shared__ uint4 sB4[KT * 4 * 32 * 2];       // 64 KB B panel: [(kt*4+j)*32+lane]*2+h

    int tid  = threadIdx.x;
    int lane = tid & 31;
    int wv   = tid >> 5;
    int NT   = N >> 4;
    int nt0  = blockIdx.y * 4;

    // ---- stage B panel (full K, 4 n-tiles) into LDS --------------------
    for (int c = tid; c < KT * 4 * 32 * 2; c += 256) {
        int h  = c & 1;
        int ln = (c >> 1) & 31;
        int j  = (c >> 6) & 3;
        int kt = c >> 8;
        const uint4* g =
            (const uint4*)(Bp + (((size_t)kt * NT + nt0 + j) * 32 + ln) * 8) + h;
#if USE_ASYNC
        // LDS byte offset = low 32 bits of the flat shared-memory address
        uint32_t ldsoff = (uint32_t)(uintptr_t)(&sB4[c]);
        asm volatile("global_load_async_to_lds_b128 %0, %1, off"
                     :: "v"(ldsoff), "v"(g) : "memory");
#else
        sB4[c] = *g;
#endif
    }
#if USE_ASYNC
    asm volatile("s_wait_asynccnt 0" ::: "memory");
#endif
    __syncthreads();

    // ---- compute: 2 m-tiles x 4 n-tiles per wave -----------------------
    int mt2   = blockIdx.x * 8 + wv;             // 32-row tile index
    int m0    = mt2 * 32 + (lane & 15);
    int half8 = (lane >> 4) << 3;                // A-layout K offset (0 or 8)
    const uint16_t* Arow0 = A + (size_t)m0 * K;
    const uint16_t* Arow1 = Arow0 + (size_t)16 * K;

    v8f acc[2][4];
#pragma unroll
    for (int t = 0; t < 2; ++t)
#pragma unroll
        for (int j = 0; j < 4; ++j)
#pragma unroll
            for (int i = 0; i < 8; ++i) acc[t][j][i] = 0.f;

    for (int kt = 0; kt < KT; ++kt) {
        FragBF a0, a1;
        int kb = kt * 32 + half8;
#pragma unroll
        for (int v = 0; v < 8; ++v) {
            int kk = kb + ((v & 4) << 2) + ((v & 3) << 1);   // A 16x32 bf16 layout
            a0.u[v] = *(const uint32_t*)(Arow0 + kk);
            a1.u[v] = *(const uint32_t*)(Arow1 + kk);
        }
#pragma unroll
        for (int j = 0; j < 4; ++j) {
            FragBF bfrag;
            int bi = ((kt * 4 + j) * 32 + lane) * 2;
            uint4 lo4 = sB4[bi], hi4 = sB4[bi + 1];
            bfrag.u[0] = lo4.x; bfrag.u[1] = lo4.y; bfrag.u[2] = lo4.z; bfrag.u[3] = lo4.w;
            bfrag.u[4] = hi4.x; bfrag.u[5] = hi4.y; bfrag.u[6] = hi4.z; bfrag.u[7] = hi4.w;
            acc[0][j] = __builtin_amdgcn_wmma_f32_16x16x32_bf16(
                false, a0.v, false, bfrag.v, (short)0, acc[0][j], false, false);
            acc[1][j] = __builtin_amdgcn_wmma_f32_16x16x32_bf16(
                false, a1.v, false, bfrag.v, (short)0, acc[1][j], false, false);
        }
    }

    // ---- store ---------------------------------------------------------
    int colbase  = lane & 15;
    int rowbase0 = mt2 * 32 + ((lane >> 4) << 3);    // C/D: lanes16-31 -> M+8
#pragma unroll
    for (int t = 0; t < 2; ++t) {
        int rowbase = rowbase0 + t * 16;
#pragma unroll
        for (int j = 0; j < 4; ++j) {
            int col = (nt0 + j) * 16 + colbase;
#pragma unroll
            for (int r = 0; r < 8; ++r) {
                size_t off = (size_t)(rowbase + r) * N + col;
                if constexpr (MODE == 0) {
                    outf[off] = resid[off] + acc[t][j][r] + bias[col];
                } else {
                    outb[off] = f2bf(acc[t][j][r]);
                }
            }
        }
    }
}

// ---------------------------------------------------------------------------
// 5) LayerNorm per row of x1[M][512] -> xn bf16. One wave per row.
__global__ void layernorm_kernel(const float* __restrict__ x1,
                                 const float* __restrict__ g,
                                 const float* __restrict__ be,
                                 uint16_t* __restrict__ xn) {
    int lane = threadIdx.x & 31;
    int wv   = threadIdx.x >> 5;
    int row  = blockIdx.x * 8 + wv;
    const float* r = x1 + (size_t)row * D;
    int base = lane * 16;
    float vals[16];
    float s = 0.f, sq = 0.f;
#pragma unroll
    for (int i = 0; i < 16; ++i) {
        float v = r[base + i];
        vals[i] = v; s += v; sq += v * v;
    }
#pragma unroll
    for (int off = 16; off >= 1; off >>= 1) {
        s  += __shfl_xor(s,  off, 32);
        sq += __shfl_xor(sq, off, 32);
    }
    float mean = s * (1.f / D);
    float var  = sq * (1.f / D) - mean * mean;
    float rstd = rsqrtf(var + 1e-5f);
    uint16_t* o = xn + (size_t)row * D;
#pragma unroll
    for (int i = 0; i < 16; ++i) {
        float v = (vals[i] - mean) * rstd * g[base + i] + be[base + i];
        o[base + i] = f2bf(v);
    }
}

// ---------------------------------------------------------------------------
// q = cls_token @ Wq  (shared across batch):  Q[NCLS][512]
__global__ void qproj_kernel(const float* __restrict__ cls,
                             const float* __restrict__ Wq,
                             float* __restrict__ Q) {
    int idx = blockIdx.x * 256 + threadIdx.x;   // 1024 outputs
    if (idx >= NCLS * D) return;
    int n = idx >> 9, col = idx & 511;
    float acc = 0.f;
    for (int k = 0; k < D; ++k) acc += cls[n * D + k] * Wq[(size_t)k * D + col];
    Q[idx] = acc;
}

// ---------------------------------------------------------------------------
// 7) Sigmoid cross-attention. One block per (b,h). kv bf16 [B*S][2D].
//    attn -> d_out+64 as [B][H][NCLS][S]; o -> [B][H][NCLS][HD] f32.
__global__ void attention_kernel(const uint16_t* __restrict__ kv,
                                 const float* __restrict__ Q,
                                 float* __restrict__ attn_out,
                                 float* __restrict__ o) {
    int bh = blockIdx.x;                 // B*H = 512
    int b = bh >> 4, h = bh & 15;
    int lane = threadIdx.x & 31;
    int wv   = threadIdx.x >> 5;
    __shared__ float sq[2][HD];
    __shared__ float part[8][2][HD];
    if (threadIdx.x < 64) {
        int n = threadIdx.x >> 5, hd = threadIdx.x & 31;
        sq[n][hd] = Q[n * D + h * HD + hd];
    }
    __syncthreads();
    float q0 = sq[0][lane], q1 = sq[1][lane];
    const float scale = 0.17677669529663687f;   // HD^-0.5
    const float lgS   = 7.624618986159398f;     // log(2048)
    float o0 = 0.f, o1 = 0.f;
    for (int s = wv; s < S; s += 8) {
        const uint16_t* row = kv + (size_t)(b * S + s) * (2 * D) + h * HD;
        float kf = bf2f(row[lane]);
        float vf = bf2f(row[D + lane]);
        float p0 = kf * q0, p1 = kf * q1;
#pragma unroll
        for (int off = 16; off >= 1; off >>= 1) {
            p0 += __shfl_xor(p0, off, 32);
            p1 += __shfl_xor(p1, off, 32);
        }
        float a0 = 1.f / (1.f + expf(-(p0 * scale - lgS)));
        float a1 = 1.f / (1.f + expf(-(p1 * scale - lgS)));
        if (lane == 0) {
            attn_out[((size_t)bh * 2 + 0) * S + s] = a0;
            attn_out[((size_t)bh * 2 + 1) * S + s] = a1;
        }
        o0 += a0 * vf;
        o1 += a1 * vf;
    }
    part[wv][0][lane] = o0;
    part[wv][1][lane] = o1;
    __syncthreads();
    if (threadIdx.x < 64) {
        int n = threadIdx.x >> 5, hd = threadIdx.x & 31;
        float acc = 0.f;
#pragma unroll
        for (int w = 0; w < 8; ++w) acc += part[w][n][hd];
        o[((size_t)bh * 2 + n) * HD + hd] = acc;
    }
}

// ---------------------------------------------------------------------------
// 8) Head: o@Wproj+bproj -> LN2 -> @Wc1+bc1, relu -> @Wc2+bc2 -> logits[b][n]
__global__ void head_kernel(const float* __restrict__ o,
                            const float* __restrict__ Wproj,
                            const float* __restrict__ bproj,
                            const float* __restrict__ g2,
                            const float* __restrict__ b2,
                            const float* __restrict__ Wc1,
                            const float* __restrict__ bc1,
                            const float* __restrict__ Wc2,
                            const float* __restrict__ bc2,
                            float* __restrict__ logits) {
    __shared__ float sA[NCLS * D];
    __shared__ float sB[NCLS * D];
    __shared__ float sred[4];
    int b = blockIdx.x, tid = threadIdx.x;
    // gather o[b][h][n][hd] -> sA[n][h*32+hd]
    for (int idx = tid; idx < NCLS * D; idx += 256) {
        int n = idx >> 9, dcol = idx & 511;
        int h = dcol >> 5, hd = dcol & 31;
        sA[idx] = o[(((size_t)b * H + h) * 2 + n) * HD + hd];
    }
    if (tid < 4) sred[tid] = 0.f;
    __syncthreads();
    // oproj = sA @ Wproj + bproj, plus LN stats via LDS atomics
    float op[4];
#pragma unroll
    for (int jj = 0; jj < 4; ++jj) {
        int idx = tid + jj * 256;
        int n = idx >> 9, col = idx & 511;
        float acc = bproj[col];
        for (int k = 0; k < D; ++k) acc += sA[n * D + k] * Wproj[(size_t)k * D + col];
        op[jj] = acc;
        atomicAdd(&sred[n * 2 + 0], acc);
        atomicAdd(&sred[n * 2 + 1], acc * acc);
    }
    __syncthreads();
    float m0 = sred[0] * (1.f / D), m1 = sred[2] * (1.f / D);
    float r0 = rsqrtf(sred[1] * (1.f / D) - m0 * m0 + 1e-5f);
    float r1 = rsqrtf(sred[3] * (1.f / D) - m1 * m1 + 1e-5f);
    __syncthreads();
    if (tid < 2) sred[tid] = 0.f;   // reuse for logits
#pragma unroll
    for (int jj = 0; jj < 4; ++jj) {
        int idx = tid + jj * 256;
        int n = idx >> 9, col = idx & 511;
        float mean = n ? m1 : m0, rstd = n ? r1 : r0;
        sB[idx] = (op[jj] - mean) * rstd * g2[col] + b2[col];
    }
    __syncthreads();
#pragma unroll
    for (int jj = 0; jj < 4; ++jj) {
        int idx = tid + jj * 256;
        int n = idx >> 9, col = idx & 511;
        float acc = bc1[col];
        for (int k = 0; k < D; ++k) acc += sB[n * D + k] * Wc1[(size_t)k * D + col];
        float hv = fmaxf(acc, 0.f);
        atomicAdd(&sred[n], hv * Wc2[col]);
    }
    __syncthreads();
    if (tid < 2) logits[b * 2 + tid] = sred[tid] + bc2[0];
}

// ---------------------------------------------------------------------------
extern "C" void kernel_launch(void* const* d_in, const int* in_sizes, int n_in,
                              void* d_out, int out_size, void* d_ws, size_t ws_size,
                              hipStream_t stream) {
    const float* x     = (const float*)d_in[0];
    const float* wave1 = (const float*)d_in[1];
    const float* wave2 = (const float*)d_in[2];
    const float* wave3 = (const float*)d_in[3];
    const float* Wp1   = (const float*)d_in[4];
    const float* bp1   = (const float*)d_in[5];
    const float* cls   = (const float*)d_in[6];
    const float* ln1g  = (const float*)d_in[7];
    const float* ln1b  = (const float*)d_in[8];
    const float* Wq    = (const float*)d_in[9];
    const float* Wkv   = (const float*)d_in[10];
    const float* Wproj = (const float*)d_in[11];
    const float* bproj = (const float*)d_in[12];
    const float* ln2g  = (const float*)d_in[13];
    const float* ln2b  = (const float*)d_in[14];
    const float* Wc1   = (const float*)d_in[15];
    const float* bc1   = (const float*)d_in[16];
    const float* Wc2   = (const float*)d_in[17];
    const float* bc2   = (const float*)d_in[18];

    float* out      = (float*)d_out;
    float* attn_out = out + Bsz * NCLS;   // logits first, then attn [B,H,NCLS,S]

    // workspace layout
    char* ws = (char*)d_ws;
    size_t off = 0;
    uint16_t* pos = (uint16_t*)(ws + off); off += (size_t)M * D * 2;          // 64 MB (also xn)
    float*    x1  = (float*)(ws + off);    off += (size_t)M * D * 4;          // 128 MB
    uint16_t* kvb = (uint16_t*)(ws + off); off += (size_t)M * 2 * D * 2;      // 128 MB
    float*    kc  = (float*)(ws + off);    off += 65536;
    uint32_t* wp1p = (uint32_t*)(ws + off); off += (size_t)D * D * 2;         // 512 KB
    uint32_t* wkvp = (uint32_t*)(ws + off); off += (size_t)D * 2 * D * 2;     // 1 MB
    float*    Qb  = (float*)(ws + off);    off += NCLS * D * 4;
    float*    ob  = (float*)(ws + off);    off += (size_t)Bsz * H * NCLS * HD * 4;

    // 1) wavelet kernels
    wavelet_kernel<<<(D * 19 + 255) / 256, 256, 0, stream>>>(wave1, wave2, wave3, kc);
    // 2) weight packing + q projection (independent)
    pack_b_kernel<<<(D / 2) * D / 256, 256, 0, stream>>>(Wp1, wp1p, D, D);
    pack_b_kernel<<<(D / 2) * (2 * D) / 256, 256, 0, stream>>>(Wkv, wkvp, D, 2 * D);
    qproj_kernel<<<4, 256, 0, stream>>>(cls, Wq, Qb);
    // 3) depthwise conv -> pos (bf16)
    dwconv_kernel<<<Bsz * 2 * 4, 256, 0, stream>>>(x, kc, pos);
    // 4) x1 = x + pos @ Wp1 + bp1   (WMMA bf16, 32x64 tile per wave)
    wmma_gemm_kernel<0><<<dim3(M / 256, D / 64), 256, 0, stream>>>(
        pos, wp1p, x, bp1, x1, nullptr, D);
    // 5) xn = LN(x1)  (bf16, reuses pos buffer)
    layernorm_kernel<<<M / 8, 256, 0, stream>>>(x1, ln1g, ln1b, pos);
    // 6) kv = xn @ Wkv  (WMMA bf16 -> bf16)
    wmma_gemm_kernel<1><<<dim3(M / 256, (2 * D) / 64), 256, 0, stream>>>(
        pos, wkvp, nullptr, nullptr, nullptr, kvb, 2 * D);
    // 7) sigmoid attention: attn -> d_out, o -> ob
    attention_kernel<<<Bsz * H, 256, 0, stream>>>(kvb, Qb, attn_out, ob);
    // 8) head -> logits
    head_kernel<<<Bsz, 256, 0, stream>>>(ob, Wproj, bproj, ln2g, ln2b,
                                         Wc1, bc1, Wc2, bc2, out);
}
